// BotRGCN_32495722562030
// MI455X (gfx1250) — compile-verified, matrix operand order checked by reading, not run on hardware
//
#include <hip/hip_runtime.h>
#include <hip/hip_bf16.h>

typedef __attribute__((ext_vector_type(16))) _Float16 v16h;
typedef __attribute__((ext_vector_type(8)))  float    v8f;

#define NN 100000
#define EE 3200000
#define RR 5
#define DD 128
#define MT (NN / 16)        // 6250 row tiles (exact)
#define RN (RR * NN)        // 500000 (rel,dst) bins

// ---------------------------------------------------------------------------
// Pack activation X[N,128] f32 -> per-lane contiguous f16 A fragments in the
// CDNA5 16-bit A 16x32 VGPR striping. Ah[((m*4+kt)*32+lane)] = one v16h (32B).
// ---------------------------------------------------------------------------
__global__ void pack_a(const float* __restrict__ X, v16h* __restrict__ Ah)
{
    int t = blockIdx.x * blockDim.x + threadIdx.x;       // MT*4*32 threads
    if (t >= MT * 4 * 32) return;
    int lane = t & 31;
    int kt   = (t >> 5) & 3;
    int m    = t >> 7;
    int row  = m * 16 + (lane & 15);
    int kb   = (lane >> 4) * 8;
    const float* p = X + (size_t)row * DD + kt * 32 + kb;
    v16h v;
#pragma unroll
    for (int i = 0; i < 8; ++i) {
        v[i]     = (_Float16)p[i];
        v[i + 8] = (_Float16)p[16 + i];
    }
    Ah[t] = v;
}

// ---------------------------------------------------------------------------
// Pack weight W[K,Ncols] f32 -> B fragments. Bh[((tn*KT+kt)*32+lane)].
// ---------------------------------------------------------------------------
__global__ void pack_b(const float* __restrict__ W, int ldw,
                       v16h* __restrict__ Bh, int KT, int NT)
{
    int t = blockIdx.x * blockDim.x + threadIdx.x;
    if (t >= NT * KT * 32) return;
    int lane = t & 31;
    int kt   = (t / 32) % KT;
    int tn   = t / (32 * KT);
    int col  = tn * 16 + (lane & 15);
    int k0   = kt * 32 + (lane >> 4) * 8;
    v16h v;
#pragma unroll
    for (int i = 0; i < 8; ++i) {
        v[i]     = (_Float16)W[(size_t)(k0 + i) * ldw + col];
        v[i + 8] = (_Float16)W[(size_t)(k0 + 16 + i) * ldw + col];
    }
    Bh[t] = v;
}

// ---------------------------------------------------------------------------
// One-wave WMMA GEMM, full 128-wide output row block per wave:
// 8 C accumulators; k-tiles outer, column tiles inner. A loaded once per
// k-tile (2x b128), 32 v_wmma per wave. Grid (MT), block 32.
// ---------------------------------------------------------------------------
__global__ __launch_bounds__(32)
void gemm_all(const v16h* __restrict__ Ah, const v16h* __restrict__ Bh,
              const float* __restrict__ bias, float* __restrict__ out,
              int apply_lrelu)
{
    const int lane = threadIdx.x;
    const int tm = blockIdx.x;
    v8f c[8] = {};
#pragma unroll
    for (int kt = 0; kt < 4; ++kt) {
        v16h a = Ah[((size_t)tm * 4 + kt) * 32 + lane];
#pragma unroll
        for (int tn = 0; tn < 8; ++tn) {
            v16h b = Bh[((size_t)tn * 4 + kt) * 32 + lane];
            c[tn] = __builtin_amdgcn_wmma_f32_16x16x32_f16(
                        false, a, false, b, (short)0, c[tn], false, false);
        }
    }
#pragma unroll
    for (int tn = 0; tn < 8; ++tn) {
        int col = tn * 16 + (lane & 15);
        float bv = bias ? bias[col] : 0.f;
#pragma unroll
        for (int v = 0; v < 8; ++v) {
            int gm = tm * 16 + v + 8 * (lane >> 4);
            float val = c[tn][v] + bv;
            if (apply_lrelu) val = (val > 0.f) ? val : 0.01f * val;
            out[(size_t)gm * DD + col] = val;
        }
    }
}

// ---------------------------------------------------------------------------
// Encoder GEMM (K=768): A streamed directly from f32 rows once (2 C tiles so
// des/tweet are read a single time from HBM), B pre-packed. Grid (MT).
// ---------------------------------------------------------------------------
template<int K>
__global__ __launch_bounds__(32)
void gemm_enc(const float* __restrict__ X, const v16h* __restrict__ Bh,
              const float* __restrict__ bias, float* __restrict__ out)
{
    constexpr int KT = K / 32;
    const int lane = threadIdx.x;
    const int tm = blockIdx.x;
    const int row = tm * 16 + (lane & 15);
    const int kb  = (lane >> 4) * 8;
    const float* xrow = X + (size_t)row * K;
    v8f c0 = {}, c1 = {};
    for (int kt = 0; kt < KT; ++kt) {
        const float* p = xrow + kt * 32 + kb;
        v16h a;
#pragma unroll
        for (int i = 0; i < 8; ++i) {
            a[i]     = (_Float16)p[i];
            a[i + 8] = (_Float16)p[16 + i];
        }
        v16h b0 = Bh[((size_t)0 * KT + kt) * 32 + lane];
        v16h b1 = Bh[((size_t)1 * KT + kt) * 32 + lane];
        c0 = __builtin_amdgcn_wmma_f32_16x16x32_f16(false, a, false, b0, (short)0, c0, false, false);
        c1 = __builtin_amdgcn_wmma_f32_16x16x32_f16(false, a, false, b1, (short)0, c1, false, false);
    }
#pragma unroll
    for (int v = 0; v < 8; ++v) {
        int gm = tm * 16 + v + 8 * (lane >> 4);
        int col = lane & 15;
        float v0 = c0[v] + bias[col];
        float v1 = c1[v] + bias[16 + col];
        v0 = (v0 > 0.f) ? v0 : 0.01f * v0;
        v1 = (v1 > 0.f) ? v1 : 0.01f * v1;
        out[(size_t)gm * DD + col]      = v0;
        out[(size_t)gm * DD + 16 + col] = v1;
    }
}

// ---------------------------------------------------------------------------
// Tiny-K encoders (K=6 / K=11): FLOP-trivial, pure VALU.
// ---------------------------------------------------------------------------
__global__ void enc_small(const float* __restrict__ X, int K,
                          const float* __restrict__ W, const float* __restrict__ b,
                          float* __restrict__ out)
{
    int t = blockIdx.x * blockDim.x + threadIdx.x;
    if (t >= NN * 32) return;
    int n = t >> 5, col = t & 31;
    float acc = b[col];
    const float* xr = X + (size_t)n * K;
    for (int k = 0; k < K; ++k) acc += xr[k] * W[k * 32 + col];
    acc = (acc > 0.f) ? acc : 0.01f * acc;
    out[(size_t)n * DD + col] = acc;
}

// ---------------------------------------------------------------------------
// Zero buffer (float4 grid-stride).
// ---------------------------------------------------------------------------
__global__ void zero_f4(float4* __restrict__ p, size_t n4)
{
    size_t i = (size_t)blockIdx.x * blockDim.x + threadIdx.x;
    size_t stride = (size_t)gridDim.x * blockDim.x;
    float4 z = {0.f, 0.f, 0.f, 0.f};
    for (; i < n4; i += stride) p[i] = z;
}

// ===========================================================================
// CSR build (once; reused by both RGCN layers) -- replaces 820M f32 atomics
// with 3.2M int atomics + a 3-pass prefix scan over RN=500k bins.
// ===========================================================================
__global__ void edge_count(const int* __restrict__ dst,
                           const int* __restrict__ et,
                           int* __restrict__ cnt, int E)
{
    int e = blockIdx.x * blockDim.x + threadIdx.x;
    if (e >= E) return;
    atomicAdd(&cnt[(size_t)et[e] * NN + dst[e]], 1);
}

// pass 1: per-block (1024 elems) exclusive scan + block totals
__global__ __launch_bounds__(256)
void scan1(const int* __restrict__ in, int* __restrict__ out,
           int* __restrict__ bsum, int n)
{
    __shared__ int sh[256];
    int t = threadIdx.x;
    int base = blockIdx.x * 1024 + t * 4;
    int v0 = (base + 0 < n) ? in[base + 0] : 0;
    int v1 = (base + 1 < n) ? in[base + 1] : 0;
    int v2 = (base + 2 < n) ? in[base + 2] : 0;
    int v3 = (base + 3 < n) ? in[base + 3] : 0;
    int local = v0 + v1 + v2 + v3;
    sh[t] = local;
    __syncthreads();
    for (int off = 1; off < 256; off <<= 1) {
        int x = (t >= off) ? sh[t - off] : 0;
        __syncthreads();
        sh[t] += x;
        __syncthreads();
    }
    int run = sh[t] - local;                  // exclusive prefix for this thread
    if (base + 0 < n) out[base + 0] = run;  run += v0;
    if (base + 1 < n) out[base + 1] = run;  run += v1;
    if (base + 2 < n) out[base + 2] = run;  run += v2;
    if (base + 3 < n) out[base + 3] = run;
    if (t == 255) bsum[blockIdx.x] = sh[255];
}

// pass 2: single block exclusive scan of block totals (nb <= 512)
__global__ __launch_bounds__(512)
void scan2(int* __restrict__ bsum, int nb)
{
    __shared__ int sh[512];
    int t = threadIdx.x;
    int orig = (t < nb) ? bsum[t] : 0;
    sh[t] = orig;
    __syncthreads();
    for (int off = 1; off < 512; off <<= 1) {
        int x = (t >= off) ? sh[t - off] : 0;
        __syncthreads();
        sh[t] += x;
        __syncthreads();
    }
    if (t < nb) bsum[t] = sh[t] - orig;       // exclusive
}

// pass 3: add block base, write cursor copy, set off[n] = E
__global__ void scan3(int* __restrict__ off, const int* __restrict__ bsum,
                      int* __restrict__ cursor, int n)
{
    int i = blockIdx.x * blockDim.x + threadIdx.x;
    if (i < n) {
        int v = off[i] + bsum[i >> 10];
        off[i] = v;
        cursor[i] = v;
    }
    if (i == 0) off[n] = EE;
}

__global__ void fill_perm(const int* __restrict__ src, const int* __restrict__ dst,
                          const int* __restrict__ et, int* __restrict__ cursor,
                          int* __restrict__ perm, int E)
{
    int e = blockIdx.x * blockDim.x + threadIdx.x;
    if (e >= E) return;
    int bin = et[e] * NN + dst[e];
    int pos = atomicAdd(&cursor[bin], 1);
    perm[pos] = src[e];
}

// ---------------------------------------------------------------------------
// Atomic-free gather for one relation: one wave per dst node, float4 per lane.
// xout[n,:] += mean over the node's relation-r in-edges of xw[src,:].
// xw_r (51MB) is L2-resident; perm read is wave-uniform.
// ---------------------------------------------------------------------------
__global__ __launch_bounds__(256)
void gather_rel(const int* __restrict__ off, const int* __restrict__ perm,
                const float* __restrict__ xw, float* __restrict__ xout, int rel)
{
    int w = (blockIdx.x * blockDim.x + threadIdx.x) >> 5;
    int lane = threadIdx.x & 31;
    if (w >= NN) return;
    int bin = rel * NN + w;
    int s0 = off[bin], s1 = off[bin + 1];
    float4 acc = {0.f, 0.f, 0.f, 0.f};
    for (int j = s0; j < s1; ++j) {
        int s = perm[j];
        float4 v = ((const float4*)(xw + (size_t)s * DD))[lane];
        acc.x += v.x; acc.y += v.y; acc.z += v.z; acc.w += v.w;
    }
    int deg = s1 - s0;
    float inv = (deg > 0) ? 1.f / (float)deg : 0.f;
    float4* o = (float4*)(xout + (size_t)w * DD);
    float4 ov = o[lane];
    ov.x += acc.x * inv; ov.y += acc.y * inv;
    ov.z += acc.z * inv; ov.w += acc.w * inv;
    o[lane] = ov;
}

// ---------------------------------------------------------------------------
// Final [N,128] @ [128,2] + b -> d_out. Too narrow for WMMA.
// ---------------------------------------------------------------------------
__global__ void out_proj(const float* __restrict__ x,
                         const float* __restrict__ W,
                         const float* __restrict__ b,
                         float* __restrict__ out)
{
    int n = blockIdx.x * blockDim.x + threadIdx.x;
    if (n >= NN) return;
    float a0 = b[0], a1 = b[1];
    const float* xr = x + (size_t)n * DD;
#pragma unroll 8
    for (int k = 0; k < DD; ++k) {
        float v = xr[k];
        a0 += v * W[2 * k];
        a1 += v * W[2 * k + 1];
    }
    out[2 * n]     = a0;
    out[2 * n + 1] = a1;
}

extern "C" void kernel_launch(void* const* d_in, const int* in_sizes, int n_in,
                              void* d_out, int out_size, void* d_ws, size_t ws_size,
                              hipStream_t stream)
{
    const float* des      = (const float*)d_in[0];
    const float* tweet    = (const float*)d_in[1];
    const float* num_prop = (const float*)d_in[2];
    const float* cat_prop = (const float*)d_in[3];
    const int*   eidx     = (const int*)d_in[4];     // [2,E]
    const int*   etype    = (const int*)d_in[5];
    const float* W_des = (const float*)d_in[6];  const float* b_des = (const float*)d_in[7];
    const float* W_tw  = (const float*)d_in[8];  const float* b_tw  = (const float*)d_in[9];
    const float* W_np  = (const float*)d_in[10]; const float* b_np  = (const float*)d_in[11];
    const float* W_cp  = (const float*)d_in[12]; const float* b_cp  = (const float*)d_in[13];
    const float* W_in  = (const float*)d_in[14]; const float* b_in  = (const float*)d_in[15];
    const float* W_rel = (const float*)d_in[16];
    const float* W_root= (const float*)d_in[17]; const float* b_rgcn= (const float*)d_in[18];
    const float* W_o1  = (const float*)d_in[19]; const float* b_o1  = (const float*)d_in[20];
    const float* W_o2  = (const float*)d_in[21]; const float* b_o2  = (const float*)d_in[22];

    const int* esrc = eidx;
    const int* edst = eidx + EE;

    // ---- workspace layout (32B-aligned blocks first, then ints) ------------
    const size_t ND = (size_t)NN * DD;
    char* p = (char*)d_ws;
    float* h    = (float*)p;  p += ND * 4;                 // encoder out / xw
    float* x0   = (float*)p;  p += ND * 4;
    float* x1   = (float*)p;  p += ND * 4;
    v16h* Ah    = (v16h*)p;   p += (size_t)MT * 4 * 32 * 32;
    v16h* pWdes = (v16h*)p;   p += (size_t)2 * 24 * 32 * 32;
    v16h* pWtw  = (v16h*)p;   p += (size_t)2 * 24 * 32 * 32;
    v16h* pWin  = (v16h*)p;   p += (size_t)8 * 4 * 32 * 32;
    v16h* pWrel = (v16h*)p;   p += (size_t)RR * 8 * 4 * 32 * 32;
    v16h* pWroot= (v16h*)p;   p += (size_t)8 * 4 * 32 * 32;
    v16h* pWo1  = (v16h*)p;   p += (size_t)8 * 4 * 32 * 32;
    int* cnt    = (int*)p;    p += (size_t)RN * 4;
    int* off    = (int*)p;    p += (size_t)(RN + 1) * 4;
    int* cursor = (int*)p;    p += (size_t)RN * 4;
    int* bsum   = (int*)p;    p += 1024 * 4;
    int* perm   = (int*)p;    p += (size_t)EE * 4;

    const dim3 blk32(32);
    const size_t pWsz128 = (size_t)8 * 4 * 32;   // fragments per 128x128 matrix
    const int NB = (RN + 1023) / 1024;           // scan blocks (489)

    // ---- 0. pack weights (once, tiny) --------------------------------------
    pack_b<<<6, 256, 0, stream>>>(W_des, 32, pWdes, 24, 2);
    pack_b<<<6, 256, 0, stream>>>(W_tw,  32, pWtw,  24, 2);
    pack_b<<<4, 256, 0, stream>>>(W_in,  DD, pWin,   4, 8);
    for (int r = 0; r < RR; ++r)
        pack_b<<<4, 256, 0, stream>>>(W_rel + (size_t)r * DD * DD, DD,
                                      pWrel + (size_t)r * pWsz128, 4, 8);
    pack_b<<<4, 256, 0, stream>>>(W_root, DD, pWroot, 4, 8);
    pack_b<<<4, 256, 0, stream>>>(W_o1,   DD, pWo1,   4, 8);

    // ---- 1. CSR build (edge structure; shared by both layers) --------------
    zero_f4<<<512, 256, 0, stream>>>((float4*)cnt, (size_t)RN / 4);
    edge_count<<<(EE + 255) / 256, 256, 0, stream>>>(edst, etype, cnt, EE);
    scan1<<<NB, 256, 0, stream>>>(cnt, off, bsum, RN);
    scan2<<<1, 512, 0, stream>>>(bsum, NB);
    scan3<<<(RN + 255) / 256, 256, 0, stream>>>(off, bsum, cursor, RN);
    fill_perm<<<(EE + 255) / 256, 256, 0, stream>>>(esrc, edst, etype, cursor, perm, EE);

    // ---- 2. feature encoders -> h[N,128] -----------------------------------
    gemm_enc<768><<<MT, blk32, 0, stream>>>(des,   pWdes, b_des, h + 0);
    gemm_enc<768><<<MT, blk32, 0, stream>>>(tweet, pWtw,  b_tw,  h + 32);
    enc_small<<<(NN * 32 + 255) / 256, 256, 0, stream>>>(num_prop,  6, W_np, b_np, h + 64);
    enc_small<<<(NN * 32 + 255) / 256, 256, 0, stream>>>(cat_prop, 11, W_cp, b_cp, h + 96);

    // ---- 3. x0 = lrelu(h @ W_in + b_in) ------------------------------------
    pack_a<<<(MT * 4 * 32 + 255) / 256, 256, 0, stream>>>(h, Ah);
    gemm_all<<<MT, blk32, 0, stream>>>(Ah, pWin, b_in, x0, 1);

    // ---- 4. two RGCN layers (atomic-free aggregation) ----------------------
    const float* xin  = x0;
    float*       xout = x1;
    for (int layer = 0; layer < 2; ++layer) {
        pack_a<<<(MT * 4 * 32 + 255) / 256, 256, 0, stream>>>(xin, Ah);
        gemm_all<<<MT, blk32, 0, stream>>>(Ah, pWroot, b_rgcn, xout, 0);  // root+bias init
        for (int r = 0; r < RR; ++r) {
            gemm_all<<<MT, blk32, 0, stream>>>(Ah, pWrel + (size_t)r * pWsz128,
                                               nullptr, h, 0);            // xw_r
            gather_rel<<<(NN * 32 + 255) / 256, 256, 0, stream>>>(off, perm, h, xout, r);
        }
        const float* t = xin; xin = xout; xout = (float*)t;
    }

    // ---- 5. output head ----------------------------------------------------
    pack_a<<<(MT * 4 * 32 + 255) / 256, 256, 0, stream>>>(x0, Ah);
    gemm_all<<<MT, blk32, 0, stream>>>(Ah, pWo1, b_o1, x1, 1);
    out_proj<<<(NN + 255) / 256, 256, 0, stream>>>(x1, W_o2, b_o2, (float*)d_out);
}
